// ISTFTHeadONNX_6116033429978
// MI455X (gfx1250) — matrix operand order repaired and checked by reading, compiled, not run on hardware
//
#include <hip/hip_runtime.h>
#include <hip/hip_bf16.h>
#include <math.h>

// ---- problem constants ----
#define Bb    8
#define Lt    4096
#define Hh    512
#define NC    1026                 // N_FFT + 2
#define NCP   1152                 // NC padded to 9 * 128
#define FREQ  513
#define NFFT  1024
#define HOP   256
#define WIN   1024
#define KP    1056                 // K of fused iDFT GEMM, padded to 33*32
#define SAMPLES ((Lt - 1) * HOP + WIN)   // 1049344
#define PADT  ((WIN - HOP) / 2)          // 384
#define OUTL  (SAMPLES - 2 * PADT)       // 1048576 = 1<<20

// GEMM tiling: block 256 thr = 8 waves (4 in M x 2 in N);
// block tile 128(M) x 128(N) x 32(K); wave tile 32 x 64 (8 WMMAs/step)
#define BM 128
#define BN 128
#define BK 32

typedef __attribute__((ext_vector_type(16))) __bf16 v16bf;
typedef __attribute__((ext_vector_type(8)))  __bf16 v8bf;
typedef __attribute__((ext_vector_type(8)))  float  v8f;
typedef __attribute__((ext_vector_type(4)))  float  v4f;

// ---------------------------------------------------------------------------
// Prep 0: x (f32) -> xb (bf16), flat copy, 8 elems/thread, vectorized
// ---------------------------------------------------------------------------
__global__ void k_prep_x(const float* __restrict__ x, __bf16* __restrict__ xb)
{
    size_t c = (size_t)blockIdx.x * 256 + threadIdx.x;     // chunk of 8
    const float* src = x + c * 8;
    v4f a = *(const v4f*)(src);
    v4f b = *(const v4f*)(src + 4);
    v8bf o;
    o[0] = (__bf16)a[0]; o[1] = (__bf16)a[1]; o[2] = (__bf16)a[2]; o[3] = (__bf16)a[3];
    o[4] = (__bf16)b[0]; o[5] = (__bf16)b[1]; o[6] = (__bf16)b[2]; o[7] = (__bf16)b[3];
    *(v8bf*)(xb + c * 8) = o;
}

// ---------------------------------------------------------------------------
// Prep 1: W_out (512 x 1026 f32) -> WpT (1152 x 512 bf16, transposed, padded)
// ---------------------------------------------------------------------------
__global__ void k_prep_w(const float* __restrict__ W, __bf16* __restrict__ WpT)
{
    int i = blockIdx.x * 256 + threadIdx.x;                // over NCP*Hh
    if (i >= NCP * Hh) return;
    int n = i >> 9;                                        // / 512
    int k = i & (Hh - 1);
    WpT[i] = (n < NC) ? (__bf16)W[(size_t)k * NC + n] : (__bf16)0.0f;
}

// ---------------------------------------------------------------------------
// Prep 2: fused iDFT basis, bf16, row-major (NFFT x KP)
// ---------------------------------------------------------------------------
__global__ void k_basis(__bf16* __restrict__ basis)
{
    int i = blockIdx.x * 256 + threadIdx.x;
    if (i >= NFFT * KP) return;
    int n = i / KP;
    int k = i - n * KP;
    float v = 0.0f;
    const float twopi = 6.28318530717958647692f;
    if (k < FREQ) {
        float sc = ((k == 0) || (k == FREQ - 1)) ? 1.0f : 2.0f;
        v = cosf(twopi * (float)n * (float)k * (1.0f / (float)NFFT)) * (sc / (float)NFFT);
    } else if (k < 2 * FREQ) {
        int f = k - FREQ;
        float sc = ((f == 0) || (f == FREQ - 1)) ? 1.0f : 2.0f;
        v = -sinf(twopi * (float)n * (float)f * (1.0f / (float)NFFT)) * (sc / (float)NFFT);
    }
    basis[i] = (__bf16)v;
}

// ---------------------------------------------------------------------------
// WMMA fragment loaders (wave32, 16x16x32 bf16)
// A frag: lane row = l16, K runs {kb..kb+7, kb+16..kb+23}, kb = 8*(lane>=16)
// B frag: lane col = l16, K runs {kb..kb+7, kb+16..kb+23}, kb = 8*(lane>=16)
// D:      vgpr r -> M = 8*(lane>=16)+r, N = l16
// ---------------------------------------------------------------------------
__device__ __forceinline__ v16bf load_a_frag(const __bf16* Asrc, int row, int kb)
{
    v8bf lo = *(const v8bf*)(Asrc + row * BK + kb);
    v8bf hi = *(const v8bf*)(Asrc + row * BK + kb + 16);
    return __builtin_shufflevector(lo, hi, 0,1,2,3,4,5,6,7,8,9,10,11,12,13,14,15);
}
__device__ __forceinline__ v16bf load_b_frag(const __bf16* Bsrc, int col, int half)
{
    v8bf lo = *(const v8bf*)(Bsrc + col * BK + half * 8);
    v8bf hi = *(const v8bf*)(Bsrc + col * BK + half * 8 + 16);
    return __builtin_shufflevector(lo, hi, 0,1,2,3,4,5,6,7,8,9,10,11,12,13,14,15);
}

// ---------------------------------------------------------------------------
// Kernel 1: h = xb @ WpT^T + b_out   (M=32768, K=512, N=1026 pad 1152) -> f32
// Double-buffered LDS, one barrier per K-step.
// ---------------------------------------------------------------------------
__global__ __launch_bounds__(256) void k_gemm1(
    const __bf16* __restrict__ xb, const __bf16* __restrict__ WpT,
    const float* __restrict__ bias, float* __restrict__ h)
{
    __shared__ __attribute__((aligned(16))) __bf16 As[2][BM * BK];
    __shared__ __attribute__((aligned(16))) __bf16 Bs[2][BN * BK];

    const int tid  = threadIdx.x;
    const int lane = tid & 31;
    const int wave = tid >> 5;
    const int mw   = wave & 3;        // 32-row slice
    const int nw   = wave >> 2;       // 64-col slice
    const int rowBase = blockIdx.x * BM;
    const int colBase = blockIdx.y * BN;

    const int half = lane >> 4;
    const int l16  = lane & 15;
    const int kb   = half * 8;

    // staging: 512 chunks of 8 per matrix; thread owns chunks tid, tid+256
    const int cm = tid >> 2;                 // 0..63
    const int ck = (tid & 3) * 8;            // 0,8,16,24

    v8f accA[4] = {{}, {}, {}, {}};
    v8f accB[4] = {{}, {}, {}, {}};
    v8bf ra0, ra1, rb0, rb1;

    const __bf16* Ag = xb  + (size_t)rowBase * Hh;
    const __bf16* Bg = WpT + (size_t)colBase * Hh;

    // preload tile 0
    ra0 = *(const v8bf*)(Ag + (size_t)cm * Hh + ck);
    ra1 = *(const v8bf*)(Ag + (size_t)(cm + 64) * Hh + ck);
    rb0 = *(const v8bf*)(Bg + (size_t)cm * Hh + ck);
    rb1 = *(const v8bf*)(Bg + (size_t)(cm + 64) * Hh + ck);
    *(v8bf*)(&As[0][cm * BK + ck])        = ra0;
    *(v8bf*)(&As[0][(cm + 64) * BK + ck]) = ra1;
    *(v8bf*)(&Bs[0][cm * BK + ck])        = rb0;
    *(v8bf*)(&Bs[0][(cm + 64) * BK + ck]) = rb1;
    __syncthreads();

    int bufi = 0;
    for (int k0 = BK; k0 < Hh; k0 += BK) {
        // issue next tile's global loads first (latency hidden by WMMAs)
        ra0 = *(const v8bf*)(Ag + (size_t)cm * Hh + k0 + ck);
        ra1 = *(const v8bf*)(Ag + (size_t)(cm + 64) * Hh + k0 + ck);
        rb0 = *(const v8bf*)(Bg + (size_t)cm * Hh + k0 + ck);
        rb1 = *(const v8bf*)(Bg + (size_t)(cm + 64) * Hh + k0 + ck);

        {
            const __bf16* Atile = As[bufi];
            const __bf16* Btile = Bs[bufi];
            v16bf a0 = load_a_frag(Atile, mw * 32 + l16, kb);
            v16bf a1 = load_a_frag(Atile, mw * 32 + 16 + l16, kb);
            #pragma unroll
            for (int s = 0; s < 4; ++s) {
                v16bf bf = load_b_frag(Btile, nw * 64 + s * 16 + l16, half);
                accA[s] = __builtin_amdgcn_wmma_f32_16x16x32_bf16(false, a0, false, bf, (short)0, accA[s], false, false);
                accB[s] = __builtin_amdgcn_wmma_f32_16x16x32_bf16(false, a1, false, bf, (short)0, accB[s], false, false);
            }
        }

        const int nb = bufi ^ 1;
        *(v8bf*)(&As[nb][cm * BK + ck])        = ra0;
        *(v8bf*)(&As[nb][(cm + 64) * BK + ck]) = ra1;
        *(v8bf*)(&Bs[nb][cm * BK + ck])        = rb0;
        *(v8bf*)(&Bs[nb][(cm + 64) * BK + ck]) = rb1;
        __syncthreads();
        bufi = nb;
    }

    // last tile
    {
        const __bf16* Atile = As[bufi];
        const __bf16* Btile = Bs[bufi];
        v16bf a0 = load_a_frag(Atile, mw * 32 + l16, kb);
        v16bf a1 = load_a_frag(Atile, mw * 32 + 16 + l16, kb);
        #pragma unroll
        for (int s = 0; s < 4; ++s) {
            v16bf bf = load_b_frag(Btile, nw * 64 + s * 16 + l16, half);
            accA[s] = __builtin_amdgcn_wmma_f32_16x16x32_bf16(false, a0, false, bf, (short)0, accA[s], false, false);
            accB[s] = __builtin_amdgcn_wmma_f32_16x16x32_bf16(false, a1, false, bf, (short)0, accB[s], false, false);
        }
    }

    // epilogue
    #pragma unroll
    for (int p = 0; p < 2; ++p) {
        int rb = rowBase + mw * 32 + p * 16 + half * 8;
        #pragma unroll
        for (int s = 0; s < 4; ++s) {
            int col = colBase + nw * 64 + s * 16 + l16;
            if (col < NC) {
                float bv = bias[col];
                v8f acc = p ? accB[s] : accA[s];
                #pragma unroll
                for (int r = 0; r < 8; ++r)
                    h[(size_t)(rb + r) * NC + col] = acc[r] + bv;
            }
        }
    }
}

// ---------------------------------------------------------------------------
// Kernel 2: RIT[b, t, row] (bf16, t-major, KP cols):
//   row <  513 : min(exp(h_mag),100) * cos(h_phase)     (real)
//   513..1025  : min(exp(h_mag),100) * sin(h_phase)     (imag)
//   row >=1026 : 0 (K padding)
// ---------------------------------------------------------------------------
__global__ void k_transform(const float* __restrict__ h, __bf16* __restrict__ RIT)
{
    size_t i = (size_t)blockIdx.x * 256 + threadIdx.x;     // over Bb*Lt*KP
    const size_t total = (size_t)Bb * Lt * KP;
    if (i >= total) return;
    int    row = (int)(i % KP);
    size_t bt  = i / KP;                                   // b*Lt + t

    float v = 0.0f;
    const float* hr = h + bt * NC;
    if (row < FREQ) {
        float mag = fminf(expf(hr[row]), 100.0f);
        v = mag * cosf(hr[FREQ + row]);
    } else if (row < 2 * FREQ) {
        int f = row - FREQ;
        float mag = fminf(expf(hr[f]), 100.0f);
        v = mag * sinf(hr[FREQ + f]);
    }
    RIT[i] = (__bf16)v;
}

// ---------------------------------------------------------------------------
// Kernel 3: frames[b, t, n] = window[n] * sum_k basis[n,k] * RIT[b,t,k]
//   M = n (1024), N = t (4096), K = 1056; frames t-major for coalesced OLA.
// ---------------------------------------------------------------------------
__global__ __launch_bounds__(256) void k_gemm2(
    const __bf16* __restrict__ basis, const __bf16* __restrict__ RIT,
    const float* __restrict__ window, float* __restrict__ frames)
{
    __shared__ __attribute__((aligned(16))) __bf16 As[2][BM * BK];
    __shared__ __attribute__((aligned(16))) __bf16 Bs[2][BN * BK];

    const int tid  = threadIdx.x;
    const int lane = tid & 31;
    const int wave = tid >> 5;
    const int mw   = wave & 3;
    const int nw   = wave >> 2;
    const int rowBase = blockIdx.x * BM;           // n
    const int colBase = blockIdx.y * BN;           // t
    const int b       = blockIdx.z;

    const int half = lane >> 4;
    const int l16  = lane & 15;
    const int kb   = half * 8;

    const int cm = tid >> 2;
    const int ck = (tid & 3) * 8;

    const __bf16* Ag = basis + (size_t)rowBase * KP;
    const __bf16* Bg = RIT + (size_t)b * Lt * KP + (size_t)colBase * KP;

    v8f accA[4] = {{}, {}, {}, {}};
    v8f accB[4] = {{}, {}, {}, {}};
    v8bf ra0, ra1, rb0, rb1;

    ra0 = *(const v8bf*)(Ag + (size_t)cm * KP + ck);
    ra1 = *(const v8bf*)(Ag + (size_t)(cm + 64) * KP + ck);
    rb0 = *(const v8bf*)(Bg + (size_t)cm * KP + ck);
    rb1 = *(const v8bf*)(Bg + (size_t)(cm + 64) * KP + ck);
    *(v8bf*)(&As[0][cm * BK + ck])        = ra0;
    *(v8bf*)(&As[0][(cm + 64) * BK + ck]) = ra1;
    *(v8bf*)(&Bs[0][cm * BK + ck])        = rb0;
    *(v8bf*)(&Bs[0][(cm + 64) * BK + ck]) = rb1;
    __syncthreads();

    int bufi = 0;
    for (int k0 = BK; k0 < KP; k0 += BK) {
        ra0 = *(const v8bf*)(Ag + (size_t)cm * KP + k0 + ck);
        ra1 = *(const v8bf*)(Ag + (size_t)(cm + 64) * KP + k0 + ck);
        rb0 = *(const v8bf*)(Bg + (size_t)cm * KP + k0 + ck);
        rb1 = *(const v8bf*)(Bg + (size_t)(cm + 64) * KP + k0 + ck);

        {
            const __bf16* Atile = As[bufi];
            const __bf16* Btile = Bs[bufi];
            v16bf a0 = load_a_frag(Atile, mw * 32 + l16, kb);
            v16bf a1 = load_a_frag(Atile, mw * 32 + 16 + l16, kb);
            #pragma unroll
            for (int s = 0; s < 4; ++s) {
                v16bf bf = load_b_frag(Btile, nw * 64 + s * 16 + l16, half);
                accA[s] = __builtin_amdgcn_wmma_f32_16x16x32_bf16(false, a0, false, bf, (short)0, accA[s], false, false);
                accB[s] = __builtin_amdgcn_wmma_f32_16x16x32_bf16(false, a1, false, bf, (short)0, accB[s], false, false);
            }
        }

        const int nb = bufi ^ 1;
        *(v8bf*)(&As[nb][cm * BK + ck])        = ra0;
        *(v8bf*)(&As[nb][(cm + 64) * BK + ck]) = ra1;
        *(v8bf*)(&Bs[nb][cm * BK + ck])        = rb0;
        *(v8bf*)(&Bs[nb][(cm + 64) * BK + ck]) = rb1;
        __syncthreads();
        bufi = nb;
    }

    {
        const __bf16* Atile = As[bufi];
        const __bf16* Btile = Bs[bufi];
        v16bf a0 = load_a_frag(Atile, mw * 32 + l16, kb);
        v16bf a1 = load_a_frag(Atile, mw * 32 + 16 + l16, kb);
        #pragma unroll
        for (int s = 0; s < 4; ++s) {
            v16bf bf = load_b_frag(Btile, nw * 64 + s * 16 + l16, half);
            accA[s] = __builtin_amdgcn_wmma_f32_16x16x32_bf16(false, a0, false, bf, (short)0, accA[s], false, false);
            accB[s] = __builtin_amdgcn_wmma_f32_16x16x32_bf16(false, a1, false, bf, (short)0, accB[s], false, false);
        }
    }

    float* fb = frames + (size_t)b * Lt * NFFT;
    #pragma unroll
    for (int p = 0; p < 2; ++p) {
        int nb0 = rowBase + mw * 32 + p * 16 + half * 8;
        #pragma unroll
        for (int s = 0; s < 4; ++s) {
            v8f acc = p ? accB[s] : accA[s];
            int t = colBase + nw * 64 + s * 16 + l16;
            float* dst = fb + (size_t)t * NFFT;
            #pragma unroll
            for (int r = 0; r < 8; ++r)
                dst[nb0 + r] = acc[r] * window[nb0 + r];
        }
    }
}

// ---------------------------------------------------------------------------
// Kernel 4: overlap-add gather + env normalize + pad trim.
// ---------------------------------------------------------------------------
__global__ void k_ola(const float* __restrict__ frames,
                      const float* __restrict__ window,
                      float* __restrict__ out)
{
    size_t i = (size_t)blockIdx.x * 256 + threadIdx.x;
    if (i >= (size_t)Bb * OUTL) return;
    int s_out = (int)(i & (OUTL - 1));
    int b     = (int)(i >> 20);
    int s     = s_out + PADT;

    int tmax = s >> 8;
    if (tmax > Lt - 1) tmax = Lt - 1;
    int tmin = (s >= WIN - 1) ? ((s - (WIN - 1) + (HOP - 1)) >> 8) : 0;

    const float* fb = frames + (size_t)b * Lt * NFFT;
    float y = 0.0f, env = 0.0f;
    for (int t = tmin; t <= tmax; ++t) {
        int n = s - (t << 8);
        float w = window[n];
        y   += fb[(size_t)t * NFFT + n];
        env += w * w;
    }
    out[i] = y / (env + 1e-11f);
}

// ---------------------------------------------------------------------------
extern "C" void kernel_launch(void* const* d_in, const int* in_sizes, int n_in,
                              void* d_out, int out_size, void* d_ws, size_t ws_size,
                              hipStream_t stream)
{
    const float* x      = (const float*)d_in[0];   // (8,4096,512)
    const float* W_out  = (const float*)d_in[1];   // (512,1026)
    const float* b_out  = (const float*)d_in[2];   // (1026,)
    const float* window = (const float*)d_in[3];   // (1024,)
    float*       out    = (float*)d_out;           // (8,1048576)

    // workspace layout:
    //   [0]     h       : 32768 x 1026 f32 (134.5 MB), reused as frames (134.2 MB)
    //   [off1]  RIT     : 8 x 4096 x 1056 bf16 (69.2 MB)
    //   [off2]  basis   : 1024 x 1056 bf16 (2.2 MB)
    //   [off3]  xb      : 32768 x 512 bf16 (33.6 MB)
    //   [off4]  WpT     : 1152 x 512 bf16 (1.2 MB)
    char* ws = (char*)d_ws;
    auto  al = [](size_t v) { return (v + 255) & ~(size_t)255; };
    size_t off1 = al((size_t)(Bb * Lt) * NC * sizeof(float));
    size_t off2 = off1 + al((size_t)Bb * Lt * KP * sizeof(__bf16));
    size_t off3 = off2 + al((size_t)NFFT * KP * sizeof(__bf16));
    size_t off4 = off3 + al((size_t)(Bb * Lt) * Hh * sizeof(__bf16));

    float*  h      = (float*)ws;
    float*  frames = (float*)ws;                 // reuse after RIT built
    __bf16* RIT    = (__bf16*)(ws + off1);
    __bf16* basis  = (__bf16*)(ws + off2);
    __bf16* xb     = (__bf16*)(ws + off3);
    __bf16* WpT    = (__bf16*)(ws + off4);

    // prep
    k_prep_x<<<(unsigned)(((size_t)(Bb * Lt) * Hh / 8 + 255) / 256), 256, 0, stream>>>(x, xb);
    k_prep_w<<<(NCP * Hh + 255) / 256, 256, 0, stream>>>(W_out, WpT);
    k_basis<<<(NFFT * KP + 255) / 256, 256, 0, stream>>>(basis);

    // 1) h = x @ W + b      (256 M-blocks, 9 N-blocks of 128)
    k_gemm1<<<dim3((Bb * Lt) / BM, NCP / BN), 256, 0, stream>>>(xb, WpT, b_out, h);

    // 2) h -> RIT (bf16, t-major, K-padded)
    {
        size_t total = (size_t)Bb * Lt * KP;
        k_transform<<<(unsigned)((total + 255) / 256), 256, 0, stream>>>(h, RIT);
    }

    // 3) frames = window * (basis @ RIT^T)   (8 M-blocks, 32 N-blocks, 8 batches)
    k_gemm2<<<dim3(NFFT / BM, Lt / BN, Bb), 256, 0, stream>>>(basis, RIT, window, frames);

    // 4) overlap-add + env normalize + trim
    {
        size_t total = (size_t)Bb * OUTL;
        k_ola<<<(unsigned)(total / 256), 256, 0, stream>>>(frames, window, out);
    }
}
